// MultiHeadAttentionBlock_24919400251654
// MI455X (gfx1250) — matrix operand order repaired and checked by reading, not compiled, not run on hardware
//
#include <hip/hip_runtime.h>

#define DMODEL 1024
#define NH 16
#define DK 64
#define SEQ 2048
#define BATCH 2

typedef __attribute__((ext_vector_type(16))) __bf16 v16bf;
typedef __attribute__((ext_vector_type(8)))  __bf16 v8bf;
typedef __attribute__((ext_vector_type(8)))  float  v8f;
typedef __attribute__((ext_vector_type(4)))  float  v4f;
typedef __attribute__((ext_vector_type(4)))  unsigned int v4u;
typedef __attribute__((ext_vector_type(8)))  int v8i;
typedef __attribute__((ext_vector_type(4)))  int v4i;

#if defined(__has_builtin)
#if __has_builtin(__builtin_amdgcn_tensor_load_to_lds) && \
    __has_builtin(__builtin_amdgcn_s_wait_tensorcnt)
#define USE_TDM 1
#endif
#endif
#ifndef USE_TDM
#define USE_TDM 0
#endif

// ---------------------------------------------------------------------------
// Fragment loaders for CDNA5 wave32 WMMA; A and B share a contiguous k order (not the ISA table):
//  A (16x32 bf16): lane m=l&15 holds row m; kh=l>>4 selects K-half.
//    elements a[0..15] = A[m][k0 + 16*kh + 0..15]
//  B (32x16 bf16): lane n=l&15 holds column n; kh=l>>4 selects K-half.
//    elements b[0..15] = B[k0 + 16*kh + 0..15][n]
//  C/D (16x16 f32): lane n=l&15, half=l>>4; c[j] = C[j + 8*half][n]
// ---------------------------------------------------------------------------

__device__ __forceinline__ v16bf load_a_f32(const float* __restrict__ base,
                                            int ld, int row0, int k0) {
  const int l = threadIdx.x & 31;
  const int m = l & 15, kh = l >> 4;
  const float* p = base + (size_t)(row0 + m) * ld + k0 + 16 * kh;
  v4f x0 = *(const v4f*)(p);
  v4f x1 = *(const v4f*)(p + 4);
  v4f x2 = *(const v4f*)(p + 8);
  v4f x3 = *(const v4f*)(p + 12);
  v16bf a;
#pragma unroll
  for (int i = 0; i < 4; ++i) {
    a[i]      = (__bf16)x0[i];
    a[4 + i]  = (__bf16)x1[i];
    a[8 + i]  = (__bf16)x2[i];
    a[12 + i] = (__bf16)x3[i];
  }
  return a;
}

__device__ __forceinline__ v16bf load_b_f32(const float* __restrict__ base,
                                            int ld, int col0, int k0) {
  const int l = threadIdx.x & 31;
  const int n = l & 15, kh = l >> 4;
  const float* p = base + (size_t)(col0 + n) * ld + k0 + 16 * kh;
  v4f x0 = *(const v4f*)(p);
  v4f x1 = *(const v4f*)(p + 4);
  v4f x2 = *(const v4f*)(p + 8);
  v4f x3 = *(const v4f*)(p + 12);
  v16bf b;
#pragma unroll
  for (int i = 0; i < 4; ++i) {
    b[i]      = (__bf16)x0[i];
    b[4 + i]  = (__bf16)x1[i];
    b[8 + i]  = (__bf16)x2[i];
    b[12 + i] = (__bf16)x3[i];
  }
  return b;
}

__device__ __forceinline__ v16bf load_a_bf16(const __bf16* __restrict__ base,
                                             int ld, int row0, int k0) {
  const int l = threadIdx.x & 31;
  const int m = l & 15, kh = l >> 4;
  const __bf16* p = base + (size_t)(row0 + m) * ld + k0 + 16 * kh;
  v8bf lo = *(const v8bf*)(p);
  v8bf hi = *(const v8bf*)(p + 8);
  return __builtin_shufflevector(lo, hi, 0, 1, 2, 3, 4, 5, 6, 7,
                                 8, 9, 10, 11, 12, 13, 14, 15);
}

__device__ __forceinline__ v16bf load_b_bf16(const __bf16* __restrict__ base,
                                             size_t ld, int col0, int k0) {
  const int l = threadIdx.x & 31;
  const int n = l & 15, kh = l >> 4;
  return *(const v16bf*)(base + (size_t)(col0 + n) * ld + k0 + 16 * kh);
}

// ---------------------------------------------------------------------------
// GEMM: out[M,N] = X[M,K](f32) @ W[N,K]^T(f32), bf16 WMMA, f32 accumulate.
// mode 0: write f32 row-major to outF
// mode 1: write bf16 head-major [B,H,S,DK] to outH  (Q / K projections)
// mode 2: write bf16 transposed [B,H,DK,S] to outH  (V projection)
// One wave computes a 64x64 tile; the 8 waves of a block share one 64-row
// A panel, which is staged into LDS by the Tensor Data Mover (double
// buffered, TENSORcnt-synchronized) when available.
// ---------------------------------------------------------------------------
__global__ void __launch_bounds__(256)
gemm_xwT_kernel(const float* __restrict__ X, const float* __restrict__ W,
                float* __restrict__ outF, __bf16* __restrict__ outH,
                int mode, int M, int N, int K) {
  extern __shared__ __align__(16) char dynsmem[];
  float* Abuf = (float*)dynsmem;  // 2 x (64 x 32) f32 panels

  const int wave = threadIdx.x >> 5;
  const int tilesN = N >> 6;
  const int tile = blockIdx.x * 8 + wave;
  const int tm = tile / tilesN;
  const int tn = tile % tilesN;
  const int tmBlk = (blockIdx.x * 8) / tilesN;  // block-uniform (8 | tilesN)
  const int l = threadIdx.x & 31, n = l & 15, half = l >> 4;

  v8f acc[4][4];
#pragma unroll
  for (int mt = 0; mt < 4; ++mt)
#pragma unroll
    for (int nt = 0; nt < 4; ++nt) acc[mt][nt] = v8f{};

#if USE_TDM
  const unsigned ldsBase = __builtin_amdgcn_groupstaticsize();  // dyn-LDS off
  const int nSlices = K / 32;

  // TDM D# for one 64-row x 32-col f32 panel of X (2-D descriptor).
  auto issue_tdm = [&](int bufIdx, int k0) {
    const unsigned long long ga =
        (unsigned long long)(const void*)(X + (size_t)(tmBlk * 64) * K + k0);
    v4u g0;
    g0[0] = 1u;                                    // count=1, user descriptor
    g0[1] = ldsBase + (unsigned)bufIdx * (64u * 32u * 4u);  // lds_addr
    g0[2] = (unsigned)ga;                          // global_addr[31:0]
    g0[3] = (unsigned)((ga >> 32) & 0x01FFFFFFull) // global_addr[56:32]
            | (2u << 30);                          // type = 2 ("image")
    v8i g1;
    g1[0] = (2 << 16);                 // data_size = 4 bytes
    g1[1] = (int)((unsigned)K << 16);  // tensor_dim0[15:0]
    g1[2] = (int)((unsigned)(M & 0xFFFF) << 16);  // tensor_dim1[15:0]
    g1[3] = (32 << 16);                // tile_dim0 = 32
    g1[4] = 64;                        // tile_dim1 = 64
    g1[5] = K;                         // tensor_dim0_stride[31:0]
    g1[6] = 0;
    g1[7] = 0;
    v4i z4 = {0, 0, 0, 0};
    v8i z8 = {0, 0, 0, 0, 0, 0, 0, 0};
    __builtin_amdgcn_tensor_load_to_lds(g0, g1, z4, z4, z8, 0);
  };

  if (wave == 0) issue_tdm(0, 0);
  for (int s = 0; s < nSlices; ++s) {
    if (wave == 0) {
      if (s + 1 < nSlices) {
        issue_tdm((s + 1) & 1, (s + 1) * 32);
        __builtin_amdgcn_s_wait_tensorcnt(1);  // slice s complete
      } else {
        __builtin_amdgcn_s_wait_tensorcnt(0);
      }
    }
    __syncthreads();  // panel s visible to all waves

    const int k0 = s * 32;
    if (k0 + 32 < K)
      __builtin_prefetch(W + (size_t)(tn * 64 + n) * K + k0 + 32, 0, 1);
    const float* Acur = Abuf + (size_t)(s & 1) * (64 * 32);
    v16bf a[4], b[4];
#pragma unroll
    for (int mt = 0; mt < 4; ++mt) a[mt] = load_a_f32(Acur, 32, mt * 16, 0);
#pragma unroll
    for (int nt = 0; nt < 4; ++nt)
      b[nt] = load_b_f32(W, K, tn * 64 + nt * 16, k0);
#pragma unroll
    for (int mt = 0; mt < 4; ++mt)
#pragma unroll
      for (int nt = 0; nt < 4; ++nt)
        acc[mt][nt] = __builtin_amdgcn_wmma_f32_16x16x32_bf16(
            false, a[mt], false, b[nt], (short)0, acc[mt][nt], false, false);

    __syncthreads();  // all reads of panel s done before its buffer is reused
  }
#else
  for (int k0 = 0; k0 < K; k0 += 32) {
    if (k0 + 32 < K)
      __builtin_prefetch(X + (size_t)(tm * 64 + (l & 15)) * K + k0 + 32, 0, 1);
    v16bf a[4], b[4];
#pragma unroll
    for (int mt = 0; mt < 4; ++mt)
      a[mt] = load_a_f32(X, K, tm * 64 + mt * 16, k0);
#pragma unroll
    for (int nt = 0; nt < 4; ++nt)
      b[nt] = load_b_f32(W, K, tn * 64 + nt * 16, k0);
#pragma unroll
    for (int mt = 0; mt < 4; ++mt)
#pragma unroll
      for (int nt = 0; nt < 4; ++nt)
        acc[mt][nt] = __builtin_amdgcn_wmma_f32_16x16x32_bf16(
            false, a[mt], false, b[nt], (short)0, acc[mt][nt], false, false);
  }
#endif

#pragma unroll
  for (int mt = 0; mt < 4; ++mt)
#pragma unroll
    for (int nt = 0; nt < 4; ++nt)
#pragma unroll
      for (int j = 0; j < 8; ++j) {
        const int gr = tm * 64 + mt * 16 + j + 8 * half;  // row in [0,M)
        const int gc = tn * 64 + nt * 16 + n;             // col in [0,N)
        const float v = acc[mt][nt][j];
        if (mode == 0) {
          outF[(size_t)gr * N + gc] = v;
        } else {
          const int bb = gr >> 11, s = gr & (SEQ - 1);
          const int h = gc >> 6, dk = gc & (DK - 1);
          if (mode == 1)
            outH[(((size_t)(bb * NH + h)) * SEQ + s) * DK + dk] = (__bf16)v;
          else
            outH[(((size_t)(bb * NH + h)) * DK + dk) * SEQ + s] = (__bf16)v;
        }
      }
}

// ---------------------------------------------------------------------------
// Flash attention: block = (q-block 256 rows, head, batch); 8 waves, each
// owns 32 query rows. Online softmax over KV chunks of 32.
// ---------------------------------------------------------------------------
__global__ void __launch_bounds__(256)
attention_kernel(const __bf16* __restrict__ Qh, const __bf16* __restrict__ Kh,
                 const __bf16* __restrict__ VT, const float* __restrict__ rel_emb,
                 const unsigned char* __restrict__ mask, float* __restrict__ X2) {
  __shared__ __align__(32) float relb[2304];
  __shared__ __align__(32) __bf16 Plds[8][32 * 32];

  const int qblk = blockIdx.x;  // 0..7
  const int h = blockIdx.y;     // 0..15
  const int b = blockIdx.z;     // 0..1
  const int q0b = qblk * 256;

  // rel_emb[:, h] slice needed by this block: index d = q - k + 2047 - q0b
  for (int i = threadIdx.x; i < 2303; i += 256)
    relb[i] = rel_emb[(size_t)(q0b + i) * NH + h];
  __syncthreads();

  const int wave = threadIdx.x >> 5;
  const int q0 = q0b + wave * 32;
  const int l = threadIdx.x & 31, n = l & 15, half = l >> 4;

  const __bf16* Qbase = Qh + (size_t)(b * NH + h) * SEQ * DK;
  const __bf16* Kbase = Kh + (size_t)(b * NH + h) * SEQ * DK;
  const __bf16* Vbase = VT + (size_t)(b * NH + h) * DK * SEQ;
  const unsigned char* mbase = mask + (size_t)b * SEQ * SEQ;

  // Q fragments: 2 row-tiles x 2 K-slices, resident for whole KV loop
  v16bf qa[2][2];
#pragma unroll
  for (int mt = 0; mt < 2; ++mt)
#pragma unroll
    for (int sl = 0; sl < 2; ++sl)
      qa[mt][sl] = load_a_bf16(Qbase, DK, q0 + mt * 16, sl * 32);

  v8f o[2][4];
  float mrow[2][8], lrow[2][8];
#pragma unroll
  for (int mt = 0; mt < 2; ++mt) {
#pragma unroll
    for (int dt = 0; dt < 4; ++dt) o[mt][dt] = v8f{};
#pragma unroll
    for (int j = 0; j < 8; ++j) { mrow[mt][j] = -1e30f; lrow[mt][j] = 0.f; }
  }

  for (int kv0 = 0; kv0 < SEQ; kv0 += 32) {
    // ---- S = Q @ K^T (B-frags of K are contiguous: B[d][kj] = K[kj][d]) ----
    v16bf bk[2][2];
#pragma unroll
    for (int nt = 0; nt < 2; ++nt)
#pragma unroll
      for (int sl = 0; sl < 2; ++sl)
        bk[nt][sl] = load_b_bf16(Kbase, DK, kv0 + nt * 16, sl * 32);

    v8f sc[2][2];
#pragma unroll
    for (int mt = 0; mt < 2; ++mt)
#pragma unroll
      for (int nt = 0; nt < 2; ++nt) {
        v8f z = v8f{};
        z = __builtin_amdgcn_wmma_f32_16x16x32_bf16(false, qa[mt][0], false,
                                                    bk[nt][0], (short)0, z,
                                                    false, false);
        sc[mt][nt] = __builtin_amdgcn_wmma_f32_16x16x32_bf16(
            false, qa[mt][1], false, bk[nt][1], (short)0, z, false, false);
      }

    // ---- scale + rel-bias + mask (C layout element addressing) ----
#pragma unroll
    for (int mt = 0; mt < 2; ++mt)
#pragma unroll
      for (int nt = 0; nt < 2; ++nt)
#pragma unroll
        for (int j = 0; j < 8; ++j) {
          const int qi = q0 + mt * 16 + j + 8 * half;
          const int ki = kv0 + nt * 16 + n;
          float v = sc[mt][nt][j] * 0.125f + relb[qi - ki + 2047 - q0b];
          if (!mbase[(size_t)qi * SEQ + ki]) v = -1e9f;
          sc[mt][nt][j] = v;
        }

    // ---- online softmax ----
#pragma unroll
    for (int mt = 0; mt < 2; ++mt) {
      float scale[8];
#pragma unroll
      for (int j = 0; j < 8; ++j) {
        float rm = fmaxf(sc[mt][0][j], sc[mt][1][j]);
        rm = fmaxf(rm, __shfl_xor(rm, 1, 32));
        rm = fmaxf(rm, __shfl_xor(rm, 2, 32));
        rm = fmaxf(rm, __shfl_xor(rm, 4, 32));
        rm = fmaxf(rm, __shfl_xor(rm, 8, 32));
        const float mnew = fmaxf(mrow[mt][j], rm);
        scale[j] = __expf(mrow[mt][j] - mnew);
        mrow[mt][j] = mnew;
        const float p0 = __expf(sc[mt][0][j] - mnew);
        const float p1 = __expf(sc[mt][1][j] - mnew);
        sc[mt][0][j] = p0;
        sc[mt][1][j] = p1;
        float rs = p0 + p1;
        rs += __shfl_xor(rs, 1, 32);
        rs += __shfl_xor(rs, 2, 32);
        rs += __shfl_xor(rs, 4, 32);
        rs += __shfl_xor(rs, 8, 32);
        lrow[mt][j] = lrow[mt][j] * scale[j] + rs;
      }
#pragma unroll
      for (int dt = 0; dt < 4; ++dt)
#pragma unroll
        for (int j = 0; j < 8; ++j) o[mt][dt][j] *= scale[j];

      // P tile (C layout) -> per-wave LDS (row-major) for A-layout reload
      __bf16* pl = &Plds[wave][0];
#pragma unroll
      for (int nt = 0; nt < 2; ++nt)
#pragma unroll
        for (int j = 0; j < 8; ++j)
          pl[(mt * 16 + j + 8 * half) * 32 + nt * 16 + n] =
              (__bf16)sc[mt][nt][j];
    }

    // ---- O += P @ V  (V stored transposed so B-frags are contiguous) ----
    v16bf pa[2];
#pragma unroll
    for (int mt = 0; mt < 2; ++mt)
      pa[mt] = load_a_bf16(&Plds[wave][0], 32, mt * 16, 0);

    v16bf bv[4];
#pragma unroll
    for (int dt = 0; dt < 4; ++dt)
      bv[dt] = load_b_bf16(Vbase, SEQ, dt * 16, kv0);

#pragma unroll
    for (int mt = 0; mt < 2; ++mt)
#pragma unroll
      for (int dt = 0; dt < 4; ++dt)
        o[mt][dt] = __builtin_amdgcn_wmma_f32_16x16x32_bf16(
            false, pa[mt], false, bv[dt], (short)0, o[mt][dt], false, false);
  }

  // ---- normalize and write concat-head activations (f32) ----
#pragma unroll
  for (int mt = 0; mt < 2; ++mt) {
    float inv[8];
#pragma unroll
    for (int j = 0; j < 8; ++j) inv[j] = 1.0f / lrow[mt][j];
#pragma unroll
    for (int dt = 0; dt < 4; ++dt)
#pragma unroll
      for (int j = 0; j < 8; ++j) {
        const int qi = q0 + mt * 16 + j + 8 * half;
        const int col = h * DK + dt * 16 + n;
        X2[((size_t)(b * SEQ + qi)) * DMODEL + col] = o[mt][dt][j] * inv[j];
      }
  }
}

// ---------------------------------------------------------------------------
// Host launcher
// ---------------------------------------------------------------------------
extern "C" void kernel_launch(void* const* d_in, const int* in_sizes, int n_in,
                              void* d_out, int out_size, void* d_ws,
                              size_t ws_size, hipStream_t stream) {
  (void)in_sizes; (void)n_in; (void)out_size; (void)ws_size;
  const float* q = (const float*)d_in[0];
  const float* k = (const float*)d_in[1];
  const float* v = (const float*)d_in[2];
  const unsigned char* mask = (const unsigned char*)d_in[3];
  const float* w_q = (const float*)d_in[4];
  const float* w_k = (const float*)d_in[5];
  const float* w_v = (const float*)d_in[6];
  const float* w_o = (const float*)d_in[7];
  const float* rel_emb = (const float*)d_in[8];

  char* ws = (char*)d_ws;
  __bf16* Qh = (__bf16*)(ws);                        // 8 MB  [B,H,S,DK] bf16
  __bf16* Kh = (__bf16*)(ws + (size_t)8  * 1048576); // 8 MB  [B,H,S,DK] bf16
  __bf16* VT = (__bf16*)(ws + (size_t)16 * 1048576); // 8 MB  [B,H,DK,S] bf16
  float*  X2 = (float*) (ws + (size_t)24 * 1048576); // 16 MB [B,S,D]    f32

  const int M = BATCH * SEQ, N = DMODEL, K = DMODEL;
  const int gemm_blocks = (M / 64) * (N / 64) / 8;  // 8 waves per block
  const size_t gemm_lds = 2u * 64u * 32u * sizeof(float);  // double A panel

  gemm_xwT_kernel<<<gemm_blocks, 256, gemm_lds, stream>>>(q, w_q, nullptr, Qh,
                                                          1, M, N, K);
  gemm_xwT_kernel<<<gemm_blocks, 256, gemm_lds, stream>>>(k, w_k, nullptr, Kh,
                                                          1, M, N, K);
  gemm_xwT_kernel<<<gemm_blocks, 256, gemm_lds, stream>>>(v, w_v, nullptr, VT,
                                                          2, M, N, K);
  attention_kernel<<<dim3(8, NH, BATCH), 256, 0, stream>>>(Qh, Kh, VT, rel_emb,
                                                           mask, X2);
  gemm_xwT_kernel<<<gemm_blocks, 256, gemm_lds, stream>>>(X2, w_o,
                                                          (float*)d_out,
                                                          nullptr, 0, M, N, K);
}
